// get_model_42915313222101
// MI455X (gfx1250) — matrix operand order repaired and checked by reading, not compiled, not run on hardware
//
#include <hip/hip_runtime.h>

// Point-cloud GLG network for MI455X (gfx1250, wave32, WMMA).
// Heavy GEMMs (PointNet encoder 512x512/512x384, GLG 768->384 projections)
// run on v_wmma_f32_16x16x32_f16 with f32 accumulate; activations staged in
// LDS as f16 row-major [row][K], read in contiguous k order, weights pre-converted to
// f16 [N][K] so B fragments are contiguous 32B loads from L2.

#define B_ 16
#define N_ 8192
#define G_ 128
#define M_ 32
#define K_ 4
#define C_ 384
#define C2_ 768
#define DEPTH_ 12
#define BG_ (B_ * G_)

typedef _Float16 h16;
typedef __attribute__((ext_vector_type(16))) _Float16 v16h;
typedef __attribute__((ext_vector_type(8)))  _Float16 v8h;
typedef __attribute__((ext_vector_type(8)))  float    v8f;

__device__ __forceinline__ v8f wmma_f16(v16h a, v16h b, v8f c) {
  return __builtin_amdgcn_wmma_f32_16x16x32_f16(false, a, false, b, (short)0, c,
                                                false, false);
}

// A fragment (16x32 f16, row-major [row][K] in LDS):
// lane<16: K = k0+{0..15}; lane>=16: k0+{16..31}. row = row0 + lane%16.
__device__ __forceinline__ v16h load_a_frag(const h16* A, int ld, int row0,
                                            int k0, int lane) {
  const h16* p = A + (size_t)(row0 + (lane & 15)) * ld + k0 + ((lane & 16) ? 16 : 0);
  v8h lo = *(const v8h*)(p);
  v8h hi = *(const v8h*)(p + 8);
  return __builtin_shufflevector(lo, hi, 0, 1, 2, 3, 4, 5, 6, 7,
                                 8, 9, 10, 11, 12, 13, 14, 15);
}

// B fragment (32x16 f16). Weights stored [N][K] so a lane's 16 K-values are
// contiguous: lane<16 covers K=k0..k0+15, lane>=16 covers k0+16..k0+31.
__device__ __forceinline__ v16h load_b_frag(const h16* W, int ld, int n0,
                                            int k0, int lane) {
  const h16* p = W + (size_t)(n0 + (lane & 15)) * ld + k0 + ((lane & 16) ? 16 : 0);
  return *(const v16h*)(p);
}

// ---------------- weight f32 -> f16 conversion (layout preserved: [N][K]) ---
__global__ __launch_bounds__(256) void kConvert(
    const float* __restrict__ c2w, const float* __restrict__ c3w,
    const float* __restrict__ c4w, const float* __restrict__ glw,
    h16* __restrict__ w2h, h16* __restrict__ w3h, h16* __restrict__ w4h,
    h16* __restrict__ wgh) {
  int idx = blockIdx.x * 256 + threadIdx.x;
  const int S2 = 256 * 128, S3 = 512 * 512, S4 = 384 * 512;
  const int SG = DEPTH_ * C_ * C2_;
  if (idx < S2) w2h[idx] = (h16)c2w[idx];
  else if (idx < S2 + S3) w3h[idx - S2] = (h16)c3w[idx - S2];
  else if (idx < S2 + S3 + S4) w4h[idx - S2 - S3] = (h16)c4w[idx - S2 - S3];
  else if (idx < S2 + S3 + S4 + SG) wgh[idx - S2 - S3 - S4] = (h16)glw[idx - S2 - S3 - S4];
}

// ---------------- furthest point sampling (one block per batch) ------------
__global__ __launch_bounds__(256) void kFPS(const float* __restrict__ pts,
                                            float* __restrict__ centers) {
  __shared__ float sD[N_];
  __shared__ float sRV[256];
  __shared__ int   sRI[256];
  __shared__ int   sFar;
  const int b = blockIdx.x, t = threadIdx.x;
  const float* px = pts + (size_t)b * 3 * N_;
  const float* py = px + N_;
  const float* pz = py + N_;
  for (int i = t; i < N_; i += 256) sD[i] = 1e10f;
  if (t == 0) sFar = 0;
  __syncthreads();
  for (int it = 0; it < G_; ++it) {
    int far = sFar;
    float fx = px[far], fy = py[far], fz = pz[far];
    if (t == 0) {
      centers[((size_t)b * G_ + it) * 3 + 0] = fx;
      centers[((size_t)b * G_ + it) * 3 + 1] = fy;
      centers[((size_t)b * G_ + it) * 3 + 2] = fz;
    }
    float bv = -1.0f; int bi = 0;
    for (int i = t; i < N_; i += 256) {
      float dx = px[i] - fx, dy = py[i] - fy, dz = pz[i] - fz;
      float d = fminf(sD[i], dx * dx + dy * dy + dz * dz);
      sD[i] = d;
      if (d > bv) { bv = d; bi = i; }
    }
    sRV[t] = bv; sRI[t] = bi;
    __syncthreads();
    for (int s = 128; s > 0; s >>= 1) {
      if (t < s) {
        if (sRV[t + s] > sRV[t] ||
            (sRV[t + s] == sRV[t] && sRI[t + s] < sRI[t])) {
          sRV[t] = sRV[t + s]; sRI[t] = sRI[t + s];
        }
      }
      __syncthreads();
    }
    if (t == 0) sFar = sRI[0];
    __syncthreads();
  }
}

// ---------------- KNN(32) per group + centered gather ----------------------
__global__ __launch_bounds__(256) void kKNN(const float* __restrict__ pts,
                                            const float* __restrict__ centers,
                                            float* __restrict__ neigh) {
  __shared__ float sD[N_];
  __shared__ float sRV[256];
  __shared__ int   sRI[256];
  const int blk = blockIdx.x, t = threadIdx.x;
  const int b = blk / G_;
  const float* px = pts + (size_t)b * 3 * N_;
  const float* py = px + N_;
  const float* pz = py + N_;
  const float cx = centers[blk * 3], cy = centers[blk * 3 + 1], cz = centers[blk * 3 + 2];
  for (int i = t; i < N_; i += 256) {
    float dx = px[i] - cx, dy = py[i] - cy, dz = pz[i] - cz;
    sD[i] = dx * dx + dy * dy + dz * dz;
  }
  __syncthreads();
  for (int m = 0; m < M_; ++m) {
    float bv = 1e30f; int bi = 0;
    for (int i = t; i < N_; i += 256) {
      float d = sD[i];
      if (d < bv) { bv = d; bi = i; }
    }
    sRV[t] = bv; sRI[t] = bi;
    __syncthreads();
    for (int s = 128; s > 0; s >>= 1) {
      if (t < s) {
        if (sRV[t + s] < sRV[t] ||
            (sRV[t + s] == sRV[t] && sRI[t + s] < sRI[t])) {
          sRV[t] = sRV[t + s]; sRI[t] = sRI[t + s];
        }
      }
      __syncthreads();
    }
    if (t == 0) {
      int sel = sRI[0];
      neigh[((size_t)blk * M_ + m) * 3 + 0] = px[sel] - cx;
      neigh[((size_t)blk * M_ + m) * 3 + 1] = py[sel] - cy;
      neigh[((size_t)blk * M_ + m) * 3 + 2] = pz[sel] - cz;
      sD[sel] = 1e30f;
    }
    __syncthreads();
  }
}

// ---------------- mini-PointNet encoder, one block per group (WMMA) --------
__global__ __launch_bounds__(256) void kEncoder(
    const float* __restrict__ neigh,
    const float* __restrict__ c1w, const float* __restrict__ c1b,
    const float* __restrict__ bn1g, const float* __restrict__ bn1b,
    const float* __restrict__ c2b, const float* __restrict__ c3b,
    const float* __restrict__ bn2g, const float* __restrict__ bn2b,
    const float* __restrict__ c4b,
    const h16* __restrict__ w2h, const h16* __restrict__ w3h,
    const h16* __restrict__ w4h,
    float* __restrict__ x) {
  __shared__ __align__(32) h16 sF1[M_ * 128];   // 8 KB
  __shared__ __align__(32) h16 sF[M_ * 512];    // 32 KB, reused twice
  __shared__ float sNeigh[M_ * 3];
  const int t = threadIdx.x, lane = t & 31, wave = t >> 5;
  const int blk = blockIdx.x;
  const float rs = rsqrtf(1.0f + 1e-5f);

  if (t < M_ * 3) sNeigh[t] = neigh[(size_t)blk * (M_ * 3) + t];
  __syncthreads();

  // stage 1: [32,3] -> [32,128], bn1 + relu (scalar, K=3)
  for (int idx = t; idx < M_ * 128; idx += 256) {
    int r = idx >> 7, c = idx & 127;
    float v = c1b[c] + sNeigh[r * 3 + 0] * c1w[c * 3 + 0]
                     + sNeigh[r * 3 + 1] * c1w[c * 3 + 1]
                     + sNeigh[r * 3 + 2] * c1w[c * 3 + 2];
    v = v * (bn1g[c] * rs) + bn1b[c];
    sF1[r * 128 + c] = (h16)fmaxf(v, 0.0f);
  }
  __syncthreads();

  // stage 2: [32,128] @ W2 -> [32,256] (WMMA), write to cols 256..511 of sF
  v8f acc2[2][2] = {};
  for (int kt = 0; kt < 4; ++kt) {
    v16h a0 = load_a_frag(sF1, 128, 0, kt * 32, lane);
    v16h a1 = load_a_frag(sF1, 128, 16, kt * 32, lane);
#pragma unroll
    for (int c = 0; c < 2; ++c) {
      v16h bf = load_b_frag(w2h, 128, (wave + 8 * c) * 16, kt * 32, lane);
      acc2[0][c] = wmma_f16(a0, bf, acc2[0][c]);
      acc2[1][c] = wmma_f16(a1, bf, acc2[1][c]);
    }
  }
#pragma unroll
  for (int c = 0; c < 2; ++c) {
    int col = (wave + 8 * c) * 16 + (lane & 15);
    float bias = c2b[col];
#pragma unroll
    for (int rt = 0; rt < 2; ++rt)
#pragma unroll
      for (int r = 0; r < 8; ++r) {
        int row = rt * 16 + r + ((lane & 16) ? 8 : 0);
        sF[row * 512 + 256 + col] = (h16)(acc2[rt][c][r] + bias);
      }
  }
  __syncthreads();

  // global max pool over M, broadcast into cols 0..255 (concat [fg | f2])
  {
    float m = -1e30f;
    for (int r = 0; r < M_; ++r) m = fmaxf(m, (float)sF[r * 512 + 256 + t]);
    h16 mh = (h16)m;
    for (int r = 0; r < M_; ++r) sF[r * 512 + t] = mh;
  }
  __syncthreads();

  // stage 3: [32,512] @ W3 -> [32,512], bn2 + relu, rewrite sF in place
  v8f acc3[2][4] = {};
  for (int kt = 0; kt < 16; ++kt) {
    v16h a0 = load_a_frag(sF, 512, 0, kt * 32, lane);
    v16h a1 = load_a_frag(sF, 512, 16, kt * 32, lane);
#pragma unroll
    for (int c = 0; c < 4; ++c) {
      if (kt + 1 < 16)
        __builtin_prefetch(w3h + (size_t)((wave + 8 * c) * 16) * 512 + (kt + 1) * 32, 0, 1);
      v16h bf = load_b_frag(w3h, 512, (wave + 8 * c) * 16, kt * 32, lane);
      acc3[0][c] = wmma_f16(a0, bf, acc3[0][c]);
      acc3[1][c] = wmma_f16(a1, bf, acc3[1][c]);
    }
  }
  __syncthreads();  // all reads of sF done before overwrite
#pragma unroll
  for (int c = 0; c < 4; ++c) {
    int col = (wave + 8 * c) * 16 + (lane & 15);
    float b3 = c3b[col], sc = bn2g[col] * rs, bb = bn2b[col];
#pragma unroll
    for (int rt = 0; rt < 2; ++rt)
#pragma unroll
      for (int r = 0; r < 8; ++r) {
        int row = rt * 16 + r + ((lane & 16) ? 8 : 0);
        float v = (acc3[rt][c][r] + b3) * sc + bb;
        sF[row * 512 + col] = (h16)fmaxf(v, 0.0f);
      }
  }
  __syncthreads();

  // stage 4: [32,512] @ W4 -> [32,384], max pool over M -> tokens
  v8f acc4[2][3] = {};
  for (int kt = 0; kt < 16; ++kt) {
    v16h a0 = load_a_frag(sF, 512, 0, kt * 32, lane);
    v16h a1 = load_a_frag(sF, 512, 16, kt * 32, lane);
#pragma unroll
    for (int c = 0; c < 3; ++c) {
      v16h bf = load_b_frag(w4h, 512, (wave + 8 * c) * 16, kt * 32, lane);
      acc4[0][c] = wmma_f16(a0, bf, acc4[0][c]);
      acc4[1][c] = wmma_f16(a1, bf, acc4[1][c]);
    }
  }
#pragma unroll
  for (int c = 0; c < 3; ++c) {
    int col = (wave + 8 * c) * 16 + (lane & 15);
    float m = -1e30f;
#pragma unroll
    for (int r = 0; r < 8; ++r) {
      m = fmaxf(m, acc4[0][c][r]);
      m = fmaxf(m, acc4[1][c][r]);
    }
    m = fmaxf(m, __shfl_xor(m, 16, 32));  // fold lane halves (rows 0-7,16-23 vs 8-15,24-31)
    if (lane < 16) x[(size_t)blk * C_ + col] = m + c4b[col];
  }
}

// ---------------- pos embed: silu(center@W1)@W2 ----------------------------
__global__ __launch_bounds__(128) void kPos(const float* __restrict__ centers,
    const float* __restrict__ pe1w, const float* __restrict__ pe1b,
    const float* __restrict__ pe2w, const float* __restrict__ pe2b,
    float* __restrict__ pos) {
  __shared__ float sH[128];
  const int blk = blockIdx.x, t = threadIdx.x;
  float cx = centers[blk * 3], cy = centers[blk * 3 + 1], cz = centers[blk * 3 + 2];
  float hv = cx * pe1w[t * 3] + cy * pe1w[t * 3 + 1] + cz * pe1w[t * 3 + 2] + pe1b[t];
  sH[t] = hv / (1.0f + expf(-hv));
  __syncthreads();
  for (int c = t; c < C_; c += 128) {
    float a = pe2b[c];
    for (int k = 0; k < 128; ++k) a += sH[k] * pe2w[c * 128 + k];
    pos[(size_t)blk * C_ + c] = a;
  }
}

// ---------------- center KNN (K=4) + gaussian weights ----------------------
__global__ __launch_bounds__(128) void kCKNN(const float* __restrict__ centers,
                                             int* __restrict__ cidx,
                                             float* __restrict__ wknn) {
  __shared__ float sC[G_ * 3];
  const int b = blockIdx.x, g = threadIdx.x;
  for (int i = g; i < G_ * 3; i += 128) sC[i] = centers[(size_t)b * G_ * 3 + i];
  __syncthreads();
  float gx = sC[g * 3], gy = sC[g * 3 + 1], gz = sC[g * 3 + 2];
  float bd[K_]; int bi[K_];
  for (int k = 0; k < K_; ++k) { bd[k] = 1e30f; bi[k] = 0; }
  for (int j = 0; j < G_; ++j) {
    float dx = sC[j * 3] - gx, dy = sC[j * 3 + 1] - gy, dz = sC[j * 3 + 2] - gz;
    float d = dx * dx + dy * dy + dz * dz;
    if (d < bd[K_ - 1]) {
      int p = K_ - 1;
      while (p > 0 && bd[p - 1] > d) { bd[p] = bd[p - 1]; bi[p] = bi[p - 1]; --p; }
      bd[p] = d; bi[p] = j;
    }
  }
  for (int k = 0; k < K_; ++k) {
    cidx[((size_t)b * G_ + g) * K_ + k] = bi[k];
    wknn[((size_t)b * G_ + g) * K_ + k] = expf(-0.5f * fmaxf(bd[k], 0.0f));
  }
}

// ---------------- 384-thread row layernorm helper --------------------------
__device__ __forceinline__ void lnstats384(float v, float* sS, float* sS2,
                                           int t, float& mu, float& iv) {
  sS[t] = v; sS2[t] = v * v;
  if (t < 128) { sS[384 + t] = 0.0f; sS2[384 + t] = 0.0f; }
  __syncthreads();
  for (int s = 256; s > 0; s >>= 1) {
    if (t < s) { sS[t] += sS[t + s]; sS2[t] += sS2[t + s]; }
    __syncthreads();
  }
  mu = sS[0] * (1.0f / C_);
  float var = sS2[0] * (1.0f / C_) - mu * mu;
  iv = rsqrtf(var + 1e-5f);
  __syncthreads();
}

// x += pos; (layer0: ln0 in place); h = LN(x, norm1[i])
__global__ __launch_bounds__(384) void kL1(float* __restrict__ x,
    const float* __restrict__ pos,
    const float* __restrict__ ln0g, const float* __restrict__ ln0b,
    const float* __restrict__ n1g, const float* __restrict__ n1b,
    float* __restrict__ h, int layer) {
  __shared__ float sS[512], sS2[512];
  const int row = blockIdx.x, t = threadIdx.x;
  size_t o = (size_t)row * C_ + t;
  float v = x[o] + pos[o];
  float mu, iv;
  lnstats384(v, sS, sS2, t, mu, iv);
  if (layer == 0) {
    v = (v - mu) * iv * ln0g[t] + ln0b[t];
    lnstats384(v, sS, sS2, t, mu, iv);
  }
  x[o] = v;
  h[o] = (v - mu) * iv * n1g[t] + n1b[t];
}

// per-(b,g) partial sums of diff and diff^2 (for global torch.std)
__global__ __launch_bounds__(256) void kL2(const float* __restrict__ h,
                                           const int* __restrict__ cidx,
                                           float* __restrict__ part) {
  __shared__ float rS[256], rS2[256];
  const int row = blockIdx.x, t = threadIdx.x;
  const int b = row / G_;
  float s = 0.f, s2 = 0.f;
  for (int idx = t; idx < K_ * C_; idx += 256) {
    int k = idx / C_, c = idx % C_;
    int nb = cidx[row * K_ + k];
    float d = h[((size_t)b * G_ + nb) * C_ + c] - h[(size_t)row * C_ + c];
    s += d; s2 += d * d;
  }
  rS[t] = s; rS2[t] = s2;
  __syncthreads();
  for (int st = 128; st > 0; st >>= 1) {
    if (t < st) { rS[t] += rS[t + st]; rS2[t] += rS2[t + st]; }
    __syncthreads();
  }
  if (t == 0) { part[row * 2] = rS[0]; part[row * 2 + 1] = rS2[0]; }
}

// deterministic final reduction -> std (ddof=1)
__global__ __launch_bounds__(256) void kL2b(const float* __restrict__ part,
                                            float* __restrict__ stdbuf) {
  __shared__ float rS[256], rS2[256];
  const int t = threadIdx.x;
  float s = 0.f, s2 = 0.f;
  for (int i = t; i < BG_; i += 256) { s += part[i * 2]; s2 += part[i * 2 + 1]; }
  rS[t] = s; rS2[t] = s2;
  __syncthreads();
  for (int st = 128; st > 0; st >>= 1) {
    if (t < st) { rS[t] += rS[t + st]; rS2[t] += rS2[t + st]; }
    __syncthreads();
  }
  if (t == 0) {
    float n = (float)(BG_ * K_ * C_);
    float var = (rS2[0] - rS[0] * rS[0] / n) / (n - 1.0f);
    stdbuf[0] = sqrtf(fmaxf(var, 0.0f));
  }
}

// GLG block: gather/normalize/max-pool -> LN -> WMMA [16,768]@[768,384] -> silu residual
__global__ __launch_bounds__(256) void kGLG(
    const float* __restrict__ h, const int* __restrict__ cidx,
    const float* __restrict__ wknn, const float* __restrict__ stdbuf,
    const float* __restrict__ alpha, const float* __restrict__ beta,
    const float* __restrict__ lng, const float* __restrict__ lnb,
    const h16* __restrict__ Wl, const float* __restrict__ linb,
    float* __restrict__ x) {
  __shared__ __align__(32) h16 sA[16 * C2_];  // 24 KB
  __shared__ float sS[256], sS2[256];
  __shared__ float sMu[16], sIs[16];
  const int t = threadIdx.x, lane = t & 31, wave = t >> 5;
  const int row0 = blockIdx.x * 16;
  const float istd = 1.0f / (stdbuf[0] + 1e-5f);

  // kx = max_k( w[k] * (alpha * [diff/std | h] + beta) )
  for (int idx = t; idx < 16 * C2_; idx += 256) {
    int r = idx / C2_, c = idx % C2_;
    int row = row0 + r;
    int b = row / G_;
    float m = -1e30f;
    if (c < C_) {
      float hv = h[(size_t)row * C_ + c];
#pragma unroll
      for (int k = 0; k < K_; ++k) {
        int nb = cidx[row * K_ + k];
        float d = h[((size_t)b * G_ + nb) * C_ + c] - hv;
        float val = alpha[c] * (d * istd) + beta[c];
        m = fmaxf(m, val * wknn[row * K_ + k]);
      }
    } else {
      float hv = h[(size_t)row * C_ + (c - C_)];
      float val = alpha[c] * hv + beta[c];
#pragma unroll
      for (int k = 0; k < K_; ++k) m = fmaxf(m, val * wknn[row * K_ + k]);
    }
    sA[idx] = (h16)m;
  }
  __syncthreads();

  // per-row LN over 768 (16 threads per row)
  {
    int r = t >> 4, li = t & 15;
    float s = 0.f, s2 = 0.f;
    for (int c = li; c < C2_; c += 16) {
      float v = (float)sA[r * C2_ + c];
      s += v; s2 += v * v;
    }
    sS[t] = s; sS2[t] = s2;
    __syncthreads();
    if (li == 0) {
      float S = 0.f, S2 = 0.f;
      for (int j = 0; j < 16; ++j) { S += sS[r * 16 + j]; S2 += sS2[r * 16 + j]; }
      float mu = S / C2_;
      float var = S2 / C2_ - mu * mu;
      sMu[r] = mu; sIs[r] = rsqrtf(var + 1e-5f);
    }
    __syncthreads();
  }
  for (int idx = t; idx < 16 * C2_; idx += 256) {
    int r = idx / C2_, c = idx % C2_;
    float v = ((float)sA[idx] - sMu[r]) * sIs[r] * lng[c] + lnb[c];
    sA[idx] = (h16)v;
  }
  __syncthreads();

  // [16,768] @ Wl^T(768->384) via WMMA; wave handles col tiles {w, w+8, w+16}
  v8f acc[3] = {};
  for (int kt = 0; kt < 24; ++kt) {
    v16h a = load_a_frag(sA, C2_, 0, kt * 32, lane);
#pragma unroll
    for (int c = 0; c < 3; ++c) {
      if (kt + 1 < 24)
        __builtin_prefetch(Wl + (size_t)((wave + 8 * c) * 16) * C2_ + (kt + 1) * 32, 0, 1);
      v16h bf = load_b_frag(Wl, C2_, (wave + 8 * c) * 16, kt * 32, lane);
      acc[c] = wmma_f16(a, bf, acc[c]);
    }
  }
#pragma unroll
  for (int c = 0; c < 3; ++c) {
    int col = (wave + 8 * c) * 16 + (lane & 15);
    float bias = linb[col];
#pragma unroll
    for (int r = 0; r < 8; ++r) {
      int row = row0 + r + ((lane & 16) ? 8 : 0);
      float v = acc[c][r] + bias;
      x[(size_t)row * C_ + col] += v / (1.0f + expf(-v));  // silu residual
    }
  }
}

// fetched feature: out[b, jofs+c, g] = LN(x, normf)
__global__ __launch_bounds__(384) void kFetch(const float* __restrict__ x,
    const float* __restrict__ fg, const float* __restrict__ fb,
    float* __restrict__ out, int jofs) {
  __shared__ float sS[512], sS2[512];
  const int row = blockIdx.x, t = threadIdx.x;
  const int b = row / G_, g = row % G_;
  float v = x[(size_t)row * C_ + t];
  float mu, iv;
  lnstats384(v, sS, sS2, t, mu, iv);
  float r = (v - mu) * iv * fg[t] + fb[t];
  out[((size_t)b * (3 * C_) + jofs + t) * G_ + g] = r;
}

extern "C" void kernel_launch(void* const* d_in, const int* in_sizes, int n_in,
                              void* d_out, int out_size, void* d_ws, size_t ws_size,
                              hipStream_t stream) {
  const float* pts  = (const float*)d_in[0];
  const float* c1w  = (const float*)d_in[1];
  const float* c1b  = (const float*)d_in[2];
  const float* bn1g = (const float*)d_in[3];
  const float* bn1b = (const float*)d_in[4];
  const float* c2w  = (const float*)d_in[5];
  const float* c2b  = (const float*)d_in[6];
  const float* c3w  = (const float*)d_in[7];
  const float* c3b  = (const float*)d_in[8];
  const float* bn2g = (const float*)d_in[9];
  const float* bn2b = (const float*)d_in[10];
  const float* c4w  = (const float*)d_in[11];
  const float* c4b  = (const float*)d_in[12];
  const float* pe1w = (const float*)d_in[13];
  const float* pe1b = (const float*)d_in[14];
  const float* pe2w = (const float*)d_in[15];
  const float* pe2b = (const float*)d_in[16];
  const float* ln0g = (const float*)d_in[17];
  const float* ln0b = (const float*)d_in[18];
  const float* n1g  = (const float*)d_in[19];
  const float* n1b  = (const float*)d_in[20];
  const float* alpha = (const float*)d_in[21];
  const float* beta  = (const float*)d_in[22];
  const float* lng   = (const float*)d_in[23];
  const float* lnb   = (const float*)d_in[24];
  const float* glw   = (const float*)d_in[25];
  const float* glb   = (const float*)d_in[26];
  const float* nfg   = (const float*)d_in[27];
  const float* nfb   = (const float*)d_in[28];
  float* out = (float*)d_out;
  char* ws = (char*)d_ws;

  // workspace layout (bytes, all offsets 256-aligned); total ~18.4 MB
  float* centers = (float*)(ws + 0);          //  24576
  float* neigh   = (float*)(ws + 24576);      // 786432
  float* x       = (float*)(ws + 811008);     // 3145728
  float* pos     = (float*)(ws + 3956736);    // 3145728
  float* h       = (float*)(ws + 7102464);    // 3145728
  int*   cidx    = (int*)  (ws + 10248192);   //  32768
  float* wknn    = (float*)(ws + 10280960);   //  32768
  float* part    = (float*)(ws + 10313728);   //  16384
  float* stdbuf  = (float*)(ws + 10330112);   //    256
  h16*   w2h     = (h16*)  (ws + 10330368);   //  65536
  h16*   w3h     = (h16*)  (ws + 10395904);   // 524288
  h16*   w4h     = (h16*)  (ws + 10920192);   // 393216
  h16*   wgh     = (h16*)  (ws + 11313408);   // 7077888

  kConvert<<<15744, 256, 0, stream>>>(c2w, c3w, c4w, glw, w2h, w3h, w4h, wgh);
  kFPS<<<B_, 256, 0, stream>>>(pts, centers);
  kKNN<<<BG_, 256, 0, stream>>>(pts, centers, neigh);
  kEncoder<<<BG_, 256, 0, stream>>>(neigh, c1w, c1b, bn1g, bn1b, c2b, c3b,
                                    bn2g, bn2b, c4b, w2h, w3h, w4h, x);
  kPos<<<BG_, 128, 0, stream>>>(centers, pe1w, pe1b, pe2w, pe2b, pos);
  kCKNN<<<B_, 128, 0, stream>>>(centers, cidx, wknn);

  int jofs = 0;
  for (int i = 0; i < DEPTH_; ++i) {
    kL1<<<BG_, 384, 0, stream>>>(x, pos, ln0g, ln0b, n1g + i * C_, n1b + i * C_, h, i);
    kL2<<<BG_, 256, 0, stream>>>(h, cidx, part);
    kL2b<<<1, 256, 0, stream>>>(part, stdbuf);
    kGLG<<<BG_ / 16, 256, 0, stream>>>(h, cidx, wknn, stdbuf,
        alpha + i * C2_, beta + i * C2_, lng + i * C2_, lnb + i * C2_,
        wgh + (size_t)i * C_ * C2_, glb + i * C_, x);
    if (i == 3 || i == 7 || i == 11) {
      kFetch<<<BG_, 384, 0, stream>>>(x, nfg, nfb, out, jofs);
      jofs += C_;
    }
  }
  (void)in_sizes; (void)n_in; (void)out_size; (void)ws_size;
}
